// RingAttention_7885559955818
// MI455X (gfx1250) — matrix operand order repaired and checked by reading, not compiled, not run on hardware
//
#include <hip/hip_runtime.h>
#include <hip/hip_bf16.h>

typedef __attribute__((ext_vector_type(16))) __bf16 v16bf;
typedef __attribute__((ext_vector_type(8)))  __bf16 v8bf;
typedef __attribute__((ext_vector_type(8)))  float  v8f;
typedef __attribute__((ext_vector_type(4)))  float  v4f;

#define RG 4          // ring size
#define RB 2          // batch
#define RH 16         // heads
#define RS 1024       // local sequence
#define RD 64         // head dim
#define RBH (RB*RH)
#define KV_TOTAL (RG*RS)
#define KBLOCK 64
#define QTILE 128
#define NWAVES 8

// log2(e) folded into 1/sqrt(D): softmax in base-2 domain throughout.
#define QSCALE 0.18033688011112042f

// ---- split barrier: drain LDS only (keep global loads in flight) ----
__device__ __forceinline__ void block_sync_lds() {
  asm volatile("s_wait_dscnt 0x0\n\t"
               "s_barrier_signal -1\n\t"
               "s_barrier_wait -1" ::: "memory");
}

// ---- single-instruction VALU max (no canonicalize ops) ----
__device__ __forceinline__ float vmaxf(float a, float b) {
  float d;
  asm("v_max_num_f32 %0, %1, %2" : "=v"(d) : "v"(a), "v"(b));
  return d;
}
// ---- fused DPP + max: d = max(x, dpp(x)) in ONE VALU op ----
__device__ __forceinline__ float vmax_xor1(float x) {
  float d;
  asm("v_max_num_f32_dpp %0, %1, %1 quad_perm:[1,0,3,2] row_mask:0xf bank_mask:0xf bound_ctrl:1"
      : "=v"(d) : "v"(x));
  return d;
}
__device__ __forceinline__ float vmax_xor2(float x) {
  float d;
  asm("v_max_num_f32_dpp %0, %1, %1 quad_perm:[2,3,0,1] row_mask:0xf bank_mask:0xf bound_ctrl:1"
      : "=v"(d) : "v"(x));
  return d;
}
__device__ __forceinline__ float vmax_ror4(float x) {
  float d;
  asm("v_max_num_f32_dpp %0, %1, %1 row_ror:4 row_mask:0xf bank_mask:0xf bound_ctrl:1"
      : "=v"(d) : "v"(x));
  return d;
}
__device__ __forceinline__ float vmax_ror8(float x) {
  float d;
  asm("v_max_num_f32_dpp %0, %1, %1 row_ror:8 row_mask:0xf bank_mask:0xf bound_ctrl:1"
      : "=v"(d) : "v"(x));
  return d;
}
__device__ __forceinline__ float rowmax16(float x) {
  x = vmax_xor1(x); x = vmax_xor2(x); x = vmax_ror4(x); x = vmax_ror8(x);
  return x;
}

__device__ __forceinline__ v16bf ld_frag16(const __bf16* p) {
  v8bf lo = *(const v8bf*)p;
  v8bf hi = *(const v8bf*)(p + 8);
  return __builtin_shufflevector(lo, hi, 0,1,2,3,4,5,6,7,8,9,10,11,12,13,14,15);
}

__device__ __forceinline__ void st_pk_bf16(__bf16* dst, float a, float b) {
  union { __bf16 h[2]; unsigned u; } x;
  x.h[0] = (__bf16)a; x.h[1] = (__bf16)b;
  *(unsigned*)dst = x.u;
}

__global__ __launch_bounds__(256)
void ring_attn_fwd(const float* __restrict__ q,
                   const float* __restrict__ k,
                   const float* __restrict__ v,
                   float* __restrict__ out)
{
  __shared__ __align__(32) __bf16 Klds[2][KBLOCK][RD];        // 16 KB double-buffered
  __shared__ __align__(32) __bf16 Vt[2][RD][KBLOCK + 16];     // 20 KB double-buffered, key-permuted V^T
  __shared__ __align__(32) __bf16 Plds[NWAVES][16][KBLOCK];   // 16 KB per-wave P scratch

  const int tid  = threadIdx.x;
  const int lane = tid & 31;
  const int wv   = tid >> 5;            // 0..7
  const int half = lane >> 4;
  const int l16  = lane & 15;

  const int blk   = blockIdx.x;
  const int qtile = blk & 7;            // 8 q-tiles of 128 rows
  const int combo = blk >> 3;           // g*BH + bh
  const int bh    = combo & (RBH - 1);

  const size_t qbase = ((size_t)combo * RS + qtile * QTILE) * RD;

  // ---------------- Q fragments (A-layout), pre-scaled ----------------
  const int qrow = wv * 16 + l16;       // row within the 128-row tile
  v16bf qa[2];
  #pragma unroll
  for (int f = 0; f < 2; ++f) {
    const float* p0 = q + qbase + (size_t)qrow * RD + 32*f + (half ? 16 : 0);
    const float* p1 = p0 + 8;
    v4f a0 = *(const v4f*)p0, a1 = *(const v4f*)(p0 + 4);
    v4f b0 = *(const v4f*)p1, b1 = *(const v4f*)(p1 + 4);
    #pragma unroll
    for (int i = 0; i < 4; ++i) {
      qa[f][i]      = (__bf16)(a0[i] * QSCALE);
      qa[f][4 + i]  = (__bf16)(a1[i] * QSCALE);
      qa[f][8 + i]  = (__bf16)(b0[i] * QSCALE);
      qa[f][12 + i] = (__bf16)(b1[i] * QSCALE);
    }
  }

  v16bf onesB;
  #pragma unroll
  for (int i = 0; i < 16; ++i) onesB[i] = (__bf16)1.0f;

  float m_run[8], l_run[8];
  v8f o[4];
  #pragma unroll
  for (int r = 0; r < 8; ++r) { m_run[r] = -1e30f; l_run[r] = 0.0f; }
  #pragma unroll
  for (int t = 0; t < 4; ++t) { v8f z = {}; o[t] = z; }

  // K staging: 256 threads x 16 floats over 64x64
  const int krow = tid >> 2;            // 0..63
  const int kcol = (tid & 3) * 16;      // 0,16,32,48
  // V staging: thread -> 8 floats from each of rows (32*vh+kk, +16)
  const int kk    = tid & 15;
  const int vh    = (tid >> 4) & 1;
  const int vcol  = (tid >> 5) * 8;     // 0..56
  const int kperm = 32 * vh + 2 * kk;

  const int nkb = KV_TOTAL / KBLOCK;    // 64

  v4f rk[4], rva[2], rvb[2];
  auto load_g = [&](int kb) {
    const int gp = (kb * KBLOCK) >> 10;
    const int kl = (kb * KBLOCK) & (RS - 1);
    const size_t base = ((size_t)(gp * RBH + bh) * RS + kl) * RD;
    const float* kp = k + base + (size_t)krow * RD + kcol;
    #pragma unroll
    for (int i = 0; i < 4; ++i) rk[i] = *(const v4f*)(kp + 4 * i);
    const float* vp0 = v + base + (size_t)(32 * vh + kk) * RD + vcol;
    const float* vp1 = vp0 + 16 * RD;
    rva[0] = *(const v4f*)vp0; rva[1] = *(const v4f*)(vp0 + 4);
    rvb[0] = *(const v4f*)vp1; rvb[1] = *(const v4f*)(vp1 + 4);
  };
  auto store_lds = [&](int buf) {
    __bf16* dk = &Klds[buf][krow][kcol];
    #pragma unroll
    for (int i = 0; i < 4; ++i) {
      st_pk_bf16(dk + 4 * i,     rk[i][0], rk[i][1]);
      st_pk_bf16(dk + 4 * i + 2, rk[i][2], rk[i][3]);
    }
    #pragma unroll
    for (int i = 0; i < 8; ++i)
      st_pk_bf16(&Vt[buf][vcol + i][kperm], rva[i >> 2][i & 3], rvb[i >> 2][i & 3]);
  };

  // -------- prologue: fill buffer 0, launch loads for block 1 --------
  load_g(0);
  store_lds(0);
  load_g(1);
  block_sync_lds();

  for (int kb = 0; kb < nkb; ++kb) {
    const int cur = kb & 1;
    if (kb + 1 < nkb) store_lds(cur ^ 1);   // stage kb+1 into other buffer
    if (kb + 2 < nkb) load_g(kb + 2);       // launch kb+2 global loads

    // ------------- scores: 4 tiles x (2 WMMAs over D=64) -------------
    v8f s[4];
    #pragma unroll
    for (int st = 0; st < 4; ++st) {
      v16bf b0 = ld_frag16(&Klds[cur][st * 16 + l16][half ? 16 : 0]);
      v16bf b1 = ld_frag16(&Klds[cur][st * 16 + l16][32 + (half ? 16 : 0)]);
      v8f acc = {};
      acc = __builtin_amdgcn_wmma_f32_16x16x32_bf16(false, qa[0], false, b0, (short)0, acc, false, false);
      acc = __builtin_amdgcn_wmma_f32_16x16x32_bf16(false, qa[1], false, b1, (short)0, acc, false, false);
      s[st] = acc;
    }

    // ------------- online softmax, base-2 domain -------------
    float csc[8], pv[4][8];
    #pragma unroll
    for (int r = 0; r < 8; ++r) {
      float bm = vmaxf(vmaxf(s[0][r], s[1][r]), vmaxf(s[2][r], s[3][r]));
      bm = rowmax16(bm);
      const float mn = vmaxf(m_run[r], bm);
      #pragma unroll
      for (int st = 0; st < 4; ++st)
        pv[st][r] = __builtin_amdgcn_exp2f(s[st][r] - mn);
      csc[r] = __builtin_amdgcn_exp2f(m_run[r] - mn);
      m_run[r] = mn;
    }
    #pragma unroll
    for (int t = 0; t < 4; ++t)
      #pragma unroll
      for (int r = 0; r < 8; ++r)
        o[t][r] *= csc[r];

    // ------------- P: interleaved per 32-key half, packed b32 stores -------------
    #pragma unroll
    for (int r = 0; r < 8; ++r) {
      st_pk_bf16(&Plds[wv][r + 8 * half][2 * l16],      pv[0][r], pv[1][r]);
      st_pk_bf16(&Plds[wv][r + 8 * half][32 + 2 * l16], pv[2][r], pv[3][r]);
    }
    const v16bf pa0 = ld_frag16(&Plds[wv][l16][half ? 16 : 0]);
    const v16bf pa1 = ld_frag16(&Plds[wv][l16][32 + (half ? 16 : 0)]);

    // ------------- rowsum-WMMAs + P x V over two key halves -------------
    v8f rz = {};
    v8f rsum = __builtin_amdgcn_wmma_f32_16x16x32_bf16(false, pa0, false, onesB, (short)0, rz, false, false);
    rsum     = __builtin_amdgcn_wmma_f32_16x16x32_bf16(false, pa1, false, onesB, (short)0, rsum, false, false);
    #pragma unroll
    for (int t = 0; t < 4; ++t) {
      v16bf vb0 = ld_frag16(&Vt[cur][t * 16 + l16][half ? 16 : 0]);
      v16bf vb1 = ld_frag16(&Vt[cur][t * 16 + l16][32 + (half ? 16 : 0)]);
      o[t] = __builtin_amdgcn_wmma_f32_16x16x32_bf16(false, pa0, false, vb0, (short)0, o[t], false, false);
      o[t] = __builtin_amdgcn_wmma_f32_16x16x32_bf16(false, pa1, false, vb1, (short)0, o[t], false, false);
    }

    #pragma unroll
    for (int r = 0; r < 8; ++r)
      l_run[r] = l_run[r] * csc[r] + rsum[r];

    block_sync_lds();
  }

  // ---------------- epilogue: normalize and store fp32 ----------------
  #pragma unroll
  for (int r = 0; r < 8; ++r) {
    const float inv = 1.0f / l_run[r];
    const int row = qtile * QTILE + wv * 16 + 8 * half + r;
    float* op = out + ((size_t)combo * RS + row) * RD + l16;
    #pragma unroll
    for (int t = 0; t < 4; ++t)
      op[t * 16] = o[t][r] * inv;
  }
}

extern "C" void kernel_launch(void* const* d_in, const int* in_sizes, int n_in,
                              void* d_out, int out_size, void* d_ws, size_t ws_size,
                              hipStream_t stream) {
  const float* q = (const float*)d_in[0];
  const float* k = (const float*)d_in[1];
  const float* v = (const float*)d_in[2];
  float* out = (float*)d_out;
  (void)in_sizes; (void)n_in; (void)out_size; (void)d_ws; (void)ws_size;

  const int blocks = RG * RB * RH * (RS / QTILE);  // 128 combos * 8 q-tiles = 1024
  ring_attn_fwd<<<blocks, 256, 0, stream>>>(q, k, v, out);
}
